// LlamaAttention_35485019799911
// MI455X (gfx1250) — matrix operand order repaired and checked by reading, not compiled, not run on hardware
//
#include <hip/hip_runtime.h>
#include <hip/hip_bf16.h>

// ---------- types ----------
typedef __attribute__((ext_vector_type(16))) __bf16 v16bf;
typedef __attribute__((ext_vector_type(8)))  float  v8f;
typedef __attribute__((ext_vector_type(4)))  unsigned int v4u;
typedef __attribute__((ext_vector_type(8)))  int v8i_;
typedef __attribute__((ext_vector_type(4)))  int v4i_;

struct alignas(16) U4 { unsigned int x, y, z, w; };
struct alignas(16) F4 { float x, y, z, w; };

union Frag {              // one WMMA A/B operand: 16 bf16 = 8 VGPRs
    v16bf v;
    U4    u[2];
};

__device__ __forceinline__ v8f wmma_bf16(const Frag& a, const Frag& b, v8f c) {
    return __builtin_amdgcn_wmma_f32_16x16x32_bf16(
        /*neg_a=*/false, a.v, /*neg_b=*/false, b.v,
        /*c_mod=*/(short)0, c, /*reuse_a=*/false, /*reuse_b=*/false);
}

// ---------- Tensor Data Mover (gfx1250 TDM) ----------
#if defined(__has_builtin)
#  if __has_builtin(__builtin_amdgcn_tensor_load_to_lds) && __has_builtin(__builtin_amdgcn_s_wait_tensorcnt)
#    define HAVE_TDM 1
#  endif
#endif
#ifndef HAVE_TDM
#  define HAVE_TDM 0
#endif

#if HAVE_TDM
// 2D tile load (bf16, data_size=2B) global -> LDS with optional LDS row padding.
// pad_interval code: pad after 2^(code+1) dwords; pad_amount code: (code+1) dwords.
__device__ __forceinline__ void tdm_load_2d(unsigned lds_byte_addr,
                                            const void* gaddr,
                                            unsigned tensor_d0, unsigned tensor_d1,
                                            unsigned tile_d0,   unsigned tile_d1,
                                            unsigned d0_stride_elems,
                                            unsigned pad_interval, unsigned pad_amount)
{
    unsigned long long ga = (unsigned long long)(size_t)gaddr;
    v4u g0;
    g0.x = 1u;                                     // count=1, user descriptor
    g0.y = lds_byte_addr;                          // lds_addr
    g0.z = (unsigned)(ga & 0xffffffffu);           // global_addr lo
    g0.w = (unsigned)((ga >> 32) & 0x01ffffffu) | (2u << 30);  // addr hi | type=2
    v8i_ g1;
    g1[0] = (int)((1u << 16)                       // data_size = 2 bytes
                | (1u << 20)                       // pad_enable
                | ((pad_interval & 7u) << 22)
                | ((pad_amount & 127u) << 25));
    g1[1] = (int)((tensor_d0 & 0xffffu) << 16);                    // dim0 lo16
    g1[2] = (int)(((tensor_d0 >> 16) & 0xffffu) | ((tensor_d1 & 0xffffu) << 16));
    g1[3] = (int)(((tensor_d1 >> 16) & 0xffffu) | ((tile_d0 & 0xffffu) << 16));
    g1[4] = (int)(tile_d1 & 0xffffu);              // tile_dim1 (tile_dim2 = 0)
    g1[5] = (int)d0_stride_elems;                  // tensor_dim0_stride lo32
    g1[6] = 0;
    g1[7] = 0;
    v4i_ z4 = {0, 0, 0, 0};
#if __clang_major__ >= 23
    v8i_ z8 = {0, 0, 0, 0, 0, 0, 0, 0};
    __builtin_amdgcn_tensor_load_to_lds(g0, g1, z4, z4, z8, 0);
#else
    __builtin_amdgcn_tensor_load_to_lds(g0, g1, z4, z4, 0);
#endif
}
#endif

// =====================================================================
// NT GEMM:  C[M,N] = A[M,K] * B[N,K]^T   (B = weight, row-major, K contig)
// A is fp32 (x) or bf16 (attn output); C is bf16 (ws) or fp32 (d_out).
// Block = 256 thr = 8 waves; tile BM=128, BN=128, BK=64 (2 WMMA k-steps).
// Wave grid 2(M) x 4(N); each wave: 4x2 = eight 16x16 WMMA accumulators.
// =====================================================================
constexpr int BM = 128, BN = 128, BK = 64;
constexpr int TP = BK + 8;          // LDS row pitch (72 bf16 = 144B, 16B aligned)

template <bool A_BF16, bool OUT_BF16>
__global__ __launch_bounds__(256)
void gemm_nt(const void* __restrict__ Ap, const float* __restrict__ Bw,
             void* __restrict__ Cp, int M, int N, int K)
{
    __shared__ alignas(16) __bf16 As[BM * TP];
    __shared__ alignas(16) __bf16 Bs[BN * TP];

    const int tid  = threadIdx.x;
    const int lane = tid & 31;
    const int wv   = tid >> 5;
    const int wm   = wv & 1;          // 0..1  -> 64 rows each
    const int wn   = wv >> 1;         // 0..3  -> 32 cols each
    const int mBlk = blockIdx.x * BM;
    const int nBlk = blockIdx.y * BN;

    v8f acc[4][2];
#pragma unroll
    for (int i = 0; i < 4; ++i)
#pragma unroll
        for (int j = 0; j < 2; ++j)
            acc[i][j] = v8f{0.f, 0.f, 0.f, 0.f, 0.f, 0.f, 0.f, 0.f};

    // cooperative tile loader: 2 threads per row, 32 elements each
    const int rowL = tid >> 1;
    const int colL = (tid & 1) * 32;

    for (int k0 = 0; k0 < K; k0 += BK) {
        __syncthreads();
        // ---- stage A tile ----
        if constexpr (A_BF16) {
#if HAVE_TDM
            if (wv == 0) {
                // 128x64 bf16 tile; rows 128B -> pad 16B: interval code 4, amount code 3
                tdm_load_2d((unsigned)(size_t)&As[0],
                            (const __bf16*)Ap + (size_t)mBlk * K + k0,
                            BK, BM, BK, BM, (unsigned)K, 4u, 3u);
                __builtin_amdgcn_s_wait_tensorcnt(0);
            }
#else
            const __bf16* Ag = (const __bf16*)Ap + (size_t)(mBlk + rowL) * K + k0 + colL;
            U4* dst = (U4*)&As[rowL * TP + colL];
#pragma unroll
            for (int e = 0; e < 4; ++e) dst[e] = *(const U4*)(Ag + e * 8);
#endif
        } else {
            const float* Ag = (const float*)Ap + (size_t)(mBlk + rowL) * K + k0 + colL;
            alignas(16) __bf16 tmp[32];
#pragma unroll
            for (int e = 0; e < 8; ++e) {
                F4 f = *(const F4*)(Ag + e * 4);
                tmp[e * 4 + 0] = (__bf16)f.x; tmp[e * 4 + 1] = (__bf16)f.y;
                tmp[e * 4 + 2] = (__bf16)f.z; tmp[e * 4 + 3] = (__bf16)f.w;
            }
            U4* dst = (U4*)&As[rowL * TP + colL];
#pragma unroll
            for (int e = 0; e < 4; ++e) dst[e] = *(const U4*)&tmp[e * 8];
        }
        // ---- stage B tile (always fp32 weights -> bf16) ----
        {
            const float* Bg = Bw + (size_t)(nBlk + rowL) * K + k0 + colL;
            alignas(16) __bf16 tmp[32];
#pragma unroll
            for (int e = 0; e < 8; ++e) {
                F4 f = *(const F4*)(Bg + e * 4);
                tmp[e * 4 + 0] = (__bf16)f.x; tmp[e * 4 + 1] = (__bf16)f.y;
                tmp[e * 4 + 2] = (__bf16)f.z; tmp[e * 4 + 3] = (__bf16)f.w;
            }
            U4* dst = (U4*)&Bs[rowL * TP + colL];
#pragma unroll
            for (int e = 0; e < 4; ++e) dst[e] = *(const U4*)&tmp[e * 8];
        }
        // prefetch next weight tile into cache hierarchy
        if (k0 + BK < K) {
            __builtin_prefetch(Bw + (size_t)(nBlk + rowL) * K + k0 + BK + colL, 0, 1);
            if (!A_BF16)
                __builtin_prefetch((const float*)Ap + (size_t)(mBlk + rowL) * K + k0 + BK + colL, 0, 1);
        }
        __syncthreads();

        // ---- WMMA compute: 2 k-steps x (4 A-frags x 2 B-frags) ----
        const int mr = lane & 15;
        const int hl = lane >> 4;      // half selects K-chunk per ISA layout
#pragma unroll
        for (int ks = 0; ks < 2; ++ks) {
            const int kk = ks * 32;
            Frag a[4], b[2];
#pragma unroll
            for (int i = 0; i < 4; ++i) {
                const __bf16* base = &As[(wm * 64 + i * 16 + mr) * TP + kk];
                a[i].u[0] = *(const U4*)(base + hl * 8);        // K 0..7 / 8..15
                a[i].u[1] = *(const U4*)(base + 16 + hl * 8);   // K 16..23 / 24..31
            }
#pragma unroll
            for (int j = 0; j < 2; ++j) {
                const __bf16* base = &Bs[(wn * 32 + j * 16 + mr) * TP + kk + hl * 8];
                b[j].u[0] = *(const U4*)(base);                 // K 0..7 / 8..15
                b[j].u[1] = *(const U4*)(base + 16);            // K 16..23 / 24..31
            }
#pragma unroll
            for (int i = 0; i < 4; ++i)
#pragma unroll
                for (int j = 0; j < 2; ++j)
                    acc[i][j] = wmma_bf16(a[i], b[j], acc[i][j]);
        }
    }

    // ---- epilogue: C/D layout (VGPR r: rows r / r+8, lane = column) ----
    const int nCol = lane & 15;
    const int mOff = (lane >> 4) * 8;
#pragma unroll
    for (int i = 0; i < 4; ++i) {
        const int mBase = mBlk + wm * 64 + i * 16 + mOff;
#pragma unroll
        for (int j = 0; j < 2; ++j) {
            const int n = nBlk + wn * 32 + j * 16 + nCol;
#pragma unroll
            for (int r = 0; r < 8; ++r) {
                float val = acc[i][j][r];
                if (OUT_BF16)
                    ((__bf16*)Cp)[(size_t)(mBase + r) * N + n] = (__bf16)val;
                else
                    ((float*)Cp)[(size_t)(mBase + r) * N + n] = val;
            }
        }
    }
}

// =====================================================================
// RoPE on q,k (bf16, layout [s, h*128+d]); fp32 math.
// =====================================================================
__global__ __launch_bounds__(256)
void rope_kernel(void* qv, void* kv, int total)
{
    int idx = blockIdx.x * blockDim.x + threadIdx.x;
    if (idx >= total) return;
    __bf16* q = (__bf16*)qv;
    __bf16* k = (__bf16*)kv;
    const int d = idx & 63;
    const int h = (idx >> 6) & 31;
    const int s = idx >> 11;
    const size_t base = (size_t)s * 4096 + h * 128 + d;
    // inv_freq = 10000^(-d/64) = exp(-d * ln(10000)/64)
    const float ang = (float)s * __expf(-0.14391156516f * (float)d);
    float c, sn;
    __sincosf(ang, &sn, &c);
    float qr = (float)q[base], qi = (float)q[base + 64];
    q[base]      = (__bf16)(qr * c - qi * sn);
    q[base + 64] = (__bf16)(qi * c + qr * sn);
    float kr = (float)k[base], ki = (float)k[base + 64];
    k[base]      = (__bf16)(kr * c - ki * sn);
    k[base + 64] = (__bf16)(ki * c + kr * sn);
}

// =====================================================================
// Flash attention. Block = 8 waves; each wave owns 16 q rows; block = 128
// rows of one head. Stream keys in tiles of 32. K tile staged by TDM into
// LDS [t][d] (padded pitch); V tile transposed manually into LDS [d][t].
// Row-sum of P obtained with an extra WMMA against an all-ones fragment.
// =====================================================================
constexpr int DH  = 128;
constexpr int BTK = 32;
constexpr int KP  = DH + 8;     // K_lds pitch (272B, 16B aligned, conflict-free)
constexpr int VP  = BTK + 8;    // Vt pitch (80B)
constexpr int PP  = BTK + 8;    // P staging pitch

__global__ __launch_bounds__(256)
void attn_kernel(const void* qv, const void* kv, const void* vv,
                 void* ov, int S)
{
    __shared__ alignas(16) __bf16 Ks[BTK * KP];
    __shared__ alignas(16) __bf16 Vt[DH * VP];
    __shared__ alignas(16) __bf16 Ps[8 * 16 * PP];

    const __bf16* q = (const __bf16*)qv;
    const __bf16* k = (const __bf16*)kv;
    const __bf16* v = (const __bf16*)vv;
    __bf16*       o = (__bf16*)ov;

    const int h    = blockIdx.y;
    const int tid  = threadIdx.x;
    const int lane = tid & 31;
    const int wv   = tid >> 5;
    const int qRow = blockIdx.x * 128 + wv * 16;
    const size_t D = 4096;
    const int mr = lane & 15;
    const int hl = lane >> 4;

    // all-ones B fragment (bf16 1.0 = 0x3f80) for rowsum-by-WMMA
    Frag ones;
    ones.u[0] = U4{0x3f803f80u, 0x3f803f80u, 0x3f803f80u, 0x3f803f80u};
    ones.u[1] = U4{0x3f803f80u, 0x3f803f80u, 0x3f803f80u, 0x3f803f80u};

    // ---- load q fragments once (16 rows x 128 d -> 4 A-frags) ----
    Frag qf[4];
    {
        const __bf16* qr = q + (size_t)(qRow + mr) * D + h * DH;
#pragma unroll
        for (int c = 0; c < 4; ++c) {
            qf[c].u[0] = *(const U4*)(qr + c * 32 + hl * 8);
            qf[c].u[1] = *(const U4*)(qr + c * 32 + 16 + hl * 8);
        }
    }

    v8f oacc[8];
#pragma unroll
    for (int i = 0; i < 8; ++i)
        oacc[i] = v8f{0.f, 0.f, 0.f, 0.f, 0.f, 0.f, 0.f, 0.f};
    v8f lacc = v8f{0.f, 0.f, 0.f, 0.f, 0.f, 0.f, 0.f, 0.f};
    float mst[8];
#pragma unroll
    for (int i = 0; i < 8; ++i) mst[i] = -1e30f;

    const float scale = 0.08838834764831845f;  // 1/sqrt(128)

    for (int t0 = 0; t0 < S; t0 += BTK) {
        __syncthreads();
        // ---- stage K tile [t][d] ----
#if HAVE_TDM
        if (wv == 0) {
            // 32x128 bf16 tile; rows 256B -> pad 16B: interval code 5, amount code 3
            tdm_load_2d((unsigned)(size_t)&Ks[0],
                        k + (size_t)t0 * D + h * DH,
                        DH, (unsigned)S, DH, BTK, (unsigned)D, 5u, 3u);
            __builtin_amdgcn_s_wait_tensorcnt(0);
        }
#else
        {
            const int tr = tid >> 3;
            const int dc = (tid & 7) * 16;
            const __bf16* kg = k + (size_t)(t0 + tr) * D + h * DH + dc;
            U4* dst = (U4*)&Ks[tr * KP + dc];
            dst[0] = *(const U4*)kg;
            dst[1] = *(const U4*)(kg + 8);
        }
#endif
        // ---- stage V tile transposed [d][t] ----
        {
            const int tr = tid >> 3;
            const int dc = (tid & 7) * 16;
            const __bf16* vg = v + (size_t)(t0 + tr) * D + h * DH + dc;
            union { U4 u; __bf16 e[8]; } r0, r1;
            r0.u = *(const U4*)vg;
            r1.u = *(const U4*)(vg + 8);
#pragma unroll
            for (int e = 0; e < 8; ++e) {
                Vt[(dc + e) * VP + tr]     = r0.e[e];
                Vt[(dc + 8 + e) * VP + tr] = r1.e[e];
            }
        }
        __syncthreads();

        // ---- scores: S(16x32) = q(16x128) * K^T; two 16x16 tiles ----
        v8f s0 = v8f{0.f,0.f,0.f,0.f,0.f,0.f,0.f,0.f};
        v8f s1 = v8f{0.f,0.f,0.f,0.f,0.f,0.f,0.f,0.f};
#pragma unroll
        for (int c = 0; c < 4; ++c) {
            Frag b0, b1;
            const __bf16* base0 = &Ks[mr * KP + c * 32 + hl * 8];
            const __bf16* base1 = &Ks[(16 + mr) * KP + c * 32 + hl * 8];
            b0.u[0] = *(const U4*)(base0); b0.u[1] = *(const U4*)(base0 + 16);
            b1.u[0] = *(const U4*)(base1); b1.u[1] = *(const U4*)(base1 + 16);
            s0 = wmma_bf16(qf[c], b0, s0);
            s1 = wmma_bf16(qf[c], b1, s1);
        }

        // ---- online softmax (row r lives in vgpr r across 16 lanes) ----
        __bf16* pw = &Ps[wv * 16 * PP];
#pragma unroll
        for (int j = 0; j < 8; ++j) {
            float a0 = s0[j] * scale;
            float a1 = s1[j] * scale;
            float mt = fmaxf(a0, a1);
#pragma unroll
            for (int d = 1; d < 16; d <<= 1)
                mt = fmaxf(mt, __shfl_xor(mt, d, 32));
            float mn = fmaxf(mst[j], mt);
            float alpha = __expf(mst[j] - mn);
            float p0 = __expf(a0 - mn);
            float p1 = __expf(a1 - mn);
            mst[j] = mn;
            lacc[j] *= alpha;
#pragma unroll
            for (int dt = 0; dt < 8; ++dt)
                oacc[dt][j] *= alpha;
            // stage P into per-wave LDS (D-layout -> memory row-major)
            const int row = j + hl * 8;
            pw[row * PP + mr]      = (__bf16)p0;
            pw[row * PP + 16 + mr] = (__bf16)p1;
        }

        // ---- reload P as A-fragment (16x32 over t) ----
        Frag pf;
        {
            const __bf16* base = &pw[mr * PP];
            pf.u[0] = *(const U4*)(base + hl * 8);
            pf.u[1] = *(const U4*)(base + 16 + hl * 8);
        }

        // ---- l += P * ones ; O += P * V ----
        lacc = wmma_bf16(pf, ones, lacc);
#pragma unroll
        for (int dt = 0; dt < 8; ++dt) {
            Frag b;
            const __bf16* base = &Vt[(dt * 16 + mr) * VP + hl * 8];
            b.u[0] = *(const U4*)(base);
            b.u[1] = *(const U4*)(base + 16);
            oacc[dt] = wmma_bf16(pf, b, oacc[dt]);
        }
    }

    // ---- normalize and store bf16 attn output [s, h*128+d] ----
#pragma unroll
    for (int dt = 0; dt < 8; ++dt) {
#pragma unroll
        for (int j = 0; j < 8; ++j) {
            float val = oacc[dt][j] / lacc[j];
            const int row = qRow + j + hl * 8;
            const int col = h * DH + dt * 16 + mr;
            o[(size_t)row * D + col] = (__bf16)val;
        }
    }
}

// =====================================================================
// launch
// =====================================================================
extern "C" void kernel_launch(void* const* d_in, const int* in_sizes, int n_in,
                              void* d_out, int out_size, void* d_ws, size_t ws_size,
                              hipStream_t stream)
{
    const float* x  = (const float*)d_in[0];
    const float* Wq = (const float*)d_in[1];
    const float* Wk = (const float*)d_in[2];
    const float* Wv = (const float*)d_in[3];
    const float* Wo = (const float*)d_in[4];
    float* out = (float*)d_out;

    const int S = 2048, D = 4096;
    __bf16* qb = (__bf16*)d_ws;
    __bf16* kb = qb + (size_t)S * D;
    __bf16* vb = kb + (size_t)S * D;
    __bf16* ab = vb + (size_t)S * D;

    dim3 gg(S / BM, D / BN), bb(256);
    // q,k,v projections (fp32 A, fp32 W -> bf16 out)
    gemm_nt<false, true><<<gg, bb, 0, stream>>>(x, Wq, qb, S, D, D);
    gemm_nt<false, true><<<gg, bb, 0, stream>>>(x, Wk, kb, S, D, D);
    gemm_nt<false, true><<<gg, bb, 0, stream>>>(x, Wv, vb, S, D, D);
    // RoPE on q,k
    {
        int total = S * 32 * 64;
        rope_kernel<<<(total + 255) / 256, 256, 0, stream>>>(qb, kb, total);
    }
    // flash attention -> ab
    attn_kernel<<<dim3(S / 128, 32), bb, 0, stream>>>(qb, kb, vb, ab, S);
    // output projection (bf16 A, fp32 W -> fp32 d_out)
    gemm_nt<true, false><<<gg, bb, 0, stream>>>(ab, Wo, out, S, D, D);
}
